// MultiHeadAttention_49950469653171
// MI455X (gfx1250) — matrix operand order repaired and checked by reading, not compiled, not run on hardware
//
#include <hip/hip_runtime.h>

#define DEV __device__ __forceinline__

typedef __attribute__((ext_vector_type(4)))  __bf16 v4bf;
typedef __attribute__((ext_vector_type(8)))  __bf16 v8bf;
typedef __attribute__((ext_vector_type(16))) __bf16 v16bf;
typedef __attribute__((ext_vector_type(8)))  float  v8f;

static constexpr int    B_  = 2;
static constexpr int    T_  = 4096;
static constexpr int    C_  = 1024;
static constexpr int    H_  = 16;
static constexpr int    DH_ = 64;
static constexpr size_t QS_ = (size_t)B_ * H_ * T_ * DH_;

#if __has_builtin(__builtin_amdgcn_global_load_async_to_lds_b128) && \
    __has_builtin(__builtin_amdgcn_s_wait_asynccnt)
#define HAVE_ASYNC_LDS 1
// prototype (from hipcc diagnostic): (int4 AS1*, int4 AS3*, imm off, imm cpol)
typedef int v4i_ __attribute__((ext_vector_type(4)));
typedef __attribute__((address_space(1))) v4i_* gas_p;
typedef __attribute__((address_space(3))) v4i_* las_p;
#endif

// ---- helpers ---------------------------------------------------------------

DEV __bf16 f2bf(float f) { return (__bf16)f; }  // v_cvt_pk_bf16_f32 when paired

DEV unsigned packbf2(__bf16 a, __bf16 b) {
  return (unsigned)__builtin_bit_cast(unsigned short, a) |
         ((unsigned)__builtin_bit_cast(unsigned short, b) << 16);
}

DEV v16bf combine16(v8bf lo, v8bf hi) {
  v16bf r;
#pragma unroll
  for (int i = 0; i < 8; ++i) { r[i] = lo[i]; r[i + 8] = hi[i]; }
  return r;
}

DEV v8f wmma_bf16(v16bf a, v16bf b, v8f c) {
  return __builtin_amdgcn_wmma_f32_16x16x32_bf16(
      false, a, false, b, (short)0, c, false, false);
}

// 16-lane-half butterfly reductions on the VALU (co-executes with WMMA).
#if __has_builtin(__builtin_amdgcn_permlane16)
DEV float plx(float v, unsigned s0, unsigned s1) {
  unsigned u = __builtin_bit_cast(unsigned, v);
  unsigned r = __builtin_amdgcn_permlane16(u, u, s0, s1, false, false);
  return __builtin_bit_cast(float, r);
}
DEV float rmax16(float v) {
  v = fmaxf(v, plx(v, 0x67452301u, 0xEFCDAB89u));  // xor 1
  v = fmaxf(v, plx(v, 0x54761032u, 0xDCFE98BAu));  // xor 2
  v = fmaxf(v, plx(v, 0x32107654u, 0xBA98FEDCu));  // xor 4
  v = fmaxf(v, plx(v, 0xFEDCBA98u, 0x76543210u));  // xor 8
  return v;
}
DEV float rsum16(float v) {
  v += plx(v, 0x67452301u, 0xEFCDAB89u);
  v += plx(v, 0x54761032u, 0xDCFE98BAu);
  v += plx(v, 0x32107654u, 0xBA98FEDCu);
  v += plx(v, 0xFEDCBA98u, 0x76543210u);
  return v;
}
#else
DEV float rmax16(float v) {
  v = fmaxf(v, __shfl_xor(v, 1, 32));
  v = fmaxf(v, __shfl_xor(v, 2, 32));
  v = fmaxf(v, __shfl_xor(v, 4, 32));
  v = fmaxf(v, __shfl_xor(v, 8, 32));
  return v;
}
DEV float rsum16(float v) {
  v += __shfl_xor(v, 1, 32);
  v += __shfl_xor(v, 2, 32);
  v += __shfl_xor(v, 4, 32);
  v += __shfl_xor(v, 8, 32);
  return v;
}
#endif

// ---- kernel 1: QKV = x @ w_qkv -> [3][B][H][T][64] bf16 --------------------
// f32 sources need cvt en route, so staging stays on the register path.
// Q pre-scaled by (1/sqrt(64)) * log2(e) so attention softmax is pure exp2.

__global__ __launch_bounds__(256) void qkv_gemm_kernel(
    const float* __restrict__ x, const float* __restrict__ w,
    __bf16* __restrict__ qkv) {
  __shared__ __align__(16) __bf16 As[2][128][72];
  __shared__ __align__(16) __bf16 Bs[2][128][72];   // [n][k] transposed
  const int m0 = blockIdx.x * 128;
  const int n0 = blockIdx.y * 128;
  const int tid = threadIdx.x;
  const int lane = tid & 31;
  const int wave = tid >> 5;
  const int wm = (wave & 3) * 32;
  const int wn = (wave >> 2) * 64;
  const int l15 = lane & 15, khalf = lane >> 4;
  v8f acc[2][4] = {};
  const int ar = tid >> 1, ac = (tid & 1) * 32;
  const int bkc = (tid & 7) * 8, bnc = (tid >> 3) * 4;
  float ra[32];
  float rb[8][4];
  auto loadA = [&](int k0) {
    const float* src = x + (size_t)(m0 + ar) * C_ + k0 + ac;
    if (k0 + 64 < C_) __builtin_prefetch(src + 64, 0, 1);
#pragma unroll
    for (int i = 0; i < 32; ++i) ra[i] = src[i];
  };
  auto loadB = [&](int k0) {
    const float* src = w + (size_t)(k0 + bkc) * (3 * C_) + n0 + bnc;
#pragma unroll
    for (int j = 0; j < 8; ++j) {
      const float* sj = src + (size_t)j * (3 * C_);
#pragma unroll
      for (int i = 0; i < 4; ++i) rb[j][i] = sj[i];
    }
  };
  auto stage = [&](int buf) {
#pragma unroll
    for (int g = 0; g < 4; ++g) {
      v8bf t;
#pragma unroll
      for (int i = 0; i < 8; ++i) t[i] = f2bf(ra[g * 8 + i]);
      *(v8bf*)&As[buf][ar][ac + g * 8] = t;
    }
#pragma unroll
    for (int i = 0; i < 4; ++i) {
      v8bf t;
#pragma unroll
      for (int j = 0; j < 8; ++j) t[j] = f2bf(rb[j][i]);
      *(v8bf*)&Bs[buf][bnc + i][bkc] = t;
    }
  };
  loadA(0); loadB(0); stage(0);
  int cur = 0;
  constexpr int STEPS = C_ / 64;
  for (int s = 0; s < STEPS; ++s) {
    __syncthreads();
    if (s + 1 < STEPS) { loadA((s + 1) * 64); loadB((s + 1) * 64); }
#pragma unroll
    for (int kc = 0; kc < 2; ++kc) {
      v16bf af[2], bfv[4];   // hoist all fragments -> scheduler spreads waits
#pragma unroll
      for (int mt = 0; mt < 2; ++mt)
        af[mt] = combine16(
            *(const v8bf*)&As[cur][wm + mt * 16 + l15][kc * 32 + khalf * 8],
            *(const v8bf*)&As[cur][wm + mt * 16 + l15][kc * 32 + 16 + khalf * 8]);
#pragma unroll
      for (int j = 0; j < 4; ++j)
        bfv[j] = combine16(
            *(const v8bf*)&Bs[cur][wn + j * 16 + l15][kc * 32 + khalf * 8],
            *(const v8bf*)&Bs[cur][wn + j * 16 + l15][kc * 32 + 16 + khalf * 8]);
#pragma unroll
      for (int j = 0; j < 4; ++j)
#pragma unroll
        for (int mt = 0; mt < 2; ++mt)
          acc[mt][j] = wmma_bf16(af[mt], bfv[j], acc[mt][j]);
    }
    if (s + 1 < STEPS) { stage(cur ^ 1); cur ^= 1; }
  }
  const int n0w = n0 + wn;
  const int sel = n0w >> 10;
  const int head = (n0w >> 6) & (H_ - 1);
  const float qscale = (sel == 0) ? (0.125f * 1.44269504088896340736f) : 1.0f;
  __bf16* dst = qkv + (size_t)sel * QS_;
#pragma unroll
  for (int mt = 0; mt < 2; ++mt) {
#pragma unroll
    for (int j = 0; j < 4; ++j) {
      const int dh = j * 16 + l15;
#pragma unroll
      for (int i = 0; i < 8; ++i) {
        const int m = m0 + wm + mt * 16 + khalf * 8 + i;
        const int bb = m >> 12, t = m & (T_ - 1);
        dst[(((size_t)(bb * H_ + head)) * T_ + t) * DH_ + dh] =
            f2bf(acc[mt][j][i] * qscale);
      }
    }
  }
}

// ---- kernel 2: flash attention, 128 q-rows per WG (8 waves x 16) -----------
// K tile staged via GLOBAL_LOAD_ASYNC_TO_LDS (DMA, ASYNCcnt); V needs a
// transpose so it keeps the register path. One barrier per 32-key chunk.

__global__ __launch_bounds__(256) void attn_kernel(
    const __bf16* __restrict__ qkv, const int* __restrict__ nvp,
    __bf16* __restrict__ aout) {
  __shared__ __align__(16) __bf16 Ks[2][32][72];
  __shared__ __align__(16) __bf16 Vt[2][64][40];
  __shared__ __align__(16) __bf16 Ps[8][16][40];
  const int q0 = blockIdx.x * 128;
  const int h = blockIdx.y;
  const int b = blockIdx.z;
  const int tid = threadIdx.x, lane = tid & 31, wave = tid >> 5;
  const int l15 = lane & 15, khalf = lane >> 4;
  const int nv = *nvp;
  const size_t headoff = ((size_t)(b * H_ + h)) * T_ * DH_;
  const __bf16* Qp = qkv + headoff;
  const __bf16* Kp = qkv + QS_ + headoff;
  const __bf16* Vp = qkv + 2 * QS_ + headoff;
  const __bf16* qrowp = Qp + (size_t)(q0 + wave * 16 + l15) * DH_;
  v16bf aq[2];
#pragma unroll
  for (int c = 0; c < 2; ++c)
    aq[c] = combine16(*(const v8bf*)(qrowp + c * 32 + khalf * 8),
                      *(const v8bf*)(qrowp + c * 32 + 16 + khalf * 8));
  float mstate[8], lstate[8];
#pragma unroll
  for (int i = 0; i < 8; ++i) { mstate[i] = -__builtin_inff(); lstate[i] = 0.f; }
  v8f oacc[4] = {};
  const int kend = (q0 >= nv) ? (q0 + 128) : T_;
  const int ksr = tid >> 3, ksc = (tid & 7) * 8;         // K: 32 rows x 8 cols
  const int vkp = (tid >> 4) * 2, vdg = (tid & 15) * 4;  // V: 2 keys x 4 dh
#ifdef HAVE_ASYNC_LDS
  auto loadK = [&](int k0, int buf) {   // async DMA: global -> LDS, no VGPRs
    const __bf16* sk = Kp + (size_t)(k0 + ksr) * DH_ + ksc;
    __builtin_amdgcn_global_load_async_to_lds_b128(
        (gas_p)sk, (las_p)&Ks[buf][ksr][ksc], 0, 0);
  };
#else
  v8bf rk;
  auto loadK = [&](int k0, int) {
    rk = *(const v8bf*)(Kp + (size_t)(k0 + ksr) * DH_ + ksc);
  };
  auto stageK = [&](int buf) { *(v8bf*)&Ks[buf][ksr][ksc] = rk; };
#endif
  v4bf rv0, rv1;
  auto loadV = [&](int k0) {
    const __bf16* sv = Vp + (size_t)(k0 + vkp) * DH_ + vdg;
    if (k0 + 32 < kend) __builtin_prefetch(sv + 32 * DH_, 0, 1);
    rv0 = *(const v4bf*)sv;
    rv1 = *(const v4bf*)(sv + DH_);
  };
  auto stageV = [&](int buf) {
#pragma unroll
    for (int i = 0; i < 4; ++i)
      *(unsigned*)&Vt[buf][vdg + i][vkp] = packbf2(rv0[i], rv1[i]);
  };
  loadK(0, 0);
  loadV(0); stageV(0);
#ifndef HAVE_ASYNC_LDS
  stageK(0);
#endif
  int cur = 0;
  for (int k0 = 0; k0 < kend; k0 += 32) {
#ifdef HAVE_ASYNC_LDS
    __builtin_amdgcn_s_wait_asynccnt(0);  // own-wave DMA done, then barrier
#endif
    __syncthreads();
    const bool more = (k0 + 32 < kend);
    if (more) { loadK(k0 + 32, cur ^ 1); loadV(k0 + 32); }
    // scores: 16q x 32keys (two 16x16 tiles), K-dim = dh in 2 chunks of 32
    v8f s0 = {}, s1 = {};
#pragma unroll
    for (int c = 0; c < 2; ++c) {
      const v16bf b0 =
          combine16(*(const v8bf*)&Ks[cur][l15][c * 32 + khalf * 8],
                    *(const v8bf*)&Ks[cur][l15][c * 32 + 16 + khalf * 8]);
      const v16bf b1 =
          combine16(*(const v8bf*)&Ks[cur][16 + l15][c * 32 + khalf * 8],
                    *(const v8bf*)&Ks[cur][16 + l15][c * 32 + 16 + khalf * 8]);
      s0 = wmma_bf16(aq[c], b0, s0);
      s1 = wmma_bf16(aq[c], b1, s1);
    }
    // mask + online softmax (log2 domain; L2E folded into Q)
#pragma unroll
    for (int i = 0; i < 8; ++i) {
      const int qrow = q0 + wave * 16 + khalf * 8 + i;
      float e0 = s0[i], e1 = s1[i];
      if (qrow >= nv) {
        const int key0 = k0 + l15, key1 = key0 + 16;
        if (key0 > qrow && key0 >= nv) e0 = -__builtin_inff();
        if (key1 > qrow && key1 >= nv) e1 = -__builtin_inff();
      }
      const float r = rmax16(fmaxf(e0, e1));
      const float mn = fmaxf(mstate[i], r);
      const bool dead = (mn == -__builtin_inff());
      const float sc = dead ? 1.f : exp2f(mstate[i] - mn);
      const float p0 = dead ? 0.f : exp2f(e0 - mn);
      const float p1 = dead ? 0.f : exp2f(e1 - mn);
      lstate[i] = lstate[i] * sc + rsum16(p0 + p1);
      mstate[i] = mn;
      Ps[wave][khalf * 8 + i][l15] = f2bf(p0);
      Ps[wave][khalf * 8 + i][16 + l15] = f2bf(p1);
#pragma unroll
      for (int jd = 0; jd < 4; ++jd) oacc[jd][i] *= sc;
    }
    // wave-private P bounce: same-wave LDS ops are in-order, no barrier needed
    const v16bf pa = combine16(*(const v8bf*)&Ps[wave][l15][khalf * 8],
                               *(const v8bf*)&Ps[wave][l15][16 + khalf * 8]);
#pragma unroll
    for (int jd = 0; jd < 4; ++jd) {
      const v16bf bv =
          combine16(*(const v8bf*)&Vt[cur][jd * 16 + l15][khalf * 8],
                    *(const v8bf*)&Vt[cur][jd * 16 + l15][16 + khalf * 8]);
      oacc[jd] = wmma_bf16(pa, bv, oacc[jd]);
    }
    if (more) {
      stageV(cur ^ 1);
#ifndef HAVE_ASYNC_LDS
      stageK(cur ^ 1);
#endif
      cur ^= 1;
    }
  }
#pragma unroll
  for (int i = 0; i < 8; ++i) {
    const float inv = (lstate[i] > 0.f) ? (1.f / lstate[i]) : 0.f;
    const int qrow = q0 + wave * 16 + khalf * 8 + i;
    __bf16* dst = aout + ((size_t)b * T_ + qrow) * C_ + h * DH_;
#pragma unroll
    for (int jd = 0; jd < 4; ++jd) dst[jd * 16 + l15] = f2bf(oacc[jd][i] * inv);
  }
}

// ---- kernel 3: out = attn @ w_out (f32 out); A tile via async DMA ----------

__global__ __launch_bounds__(256) void out_gemm_kernel(
    const __bf16* __restrict__ a, const float* __restrict__ w,
    float* __restrict__ out) {
  __shared__ __align__(16) __bf16 As[2][128][72];
  __shared__ __align__(16) __bf16 Bs[2][128][72];
  const int m0 = blockIdx.x * 128, n0 = blockIdx.y * 128;
  const int tid = threadIdx.x, lane = tid & 31, wave = tid >> 5;
  const int wm = (wave & 3) * 32, wn = (wave >> 2) * 64;
  const int l15 = lane & 15, khalf = lane >> 4;
  v8f acc[2][4] = {};
  const int ar = tid >> 1, ac = (tid & 1) * 32;
  const int bkc = (tid & 7) * 8, bnc = (tid >> 3) * 4;
#ifdef HAVE_ASYNC_LDS
  auto loadA = [&](int k0, int buf) {   // bf16 copy: async DMA path
    const __bf16* src = a + (size_t)(m0 + ar) * C_ + k0 + ac;
#pragma unroll
    for (int g = 0; g < 4; ++g)
      __builtin_amdgcn_global_load_async_to_lds_b128(
          (gas_p)(src + g * 8), (las_p)&As[buf][ar][ac + g * 8], 0, 0);
  };
#else
  v8bf rab[4];
  auto loadA = [&](int k0, int) {
    const __bf16* src = a + (size_t)(m0 + ar) * C_ + k0 + ac;
#pragma unroll
    for (int g = 0; g < 4; ++g) rab[g] = *(const v8bf*)(src + g * 8);
  };
  auto stageA = [&](int buf) {
#pragma unroll
    for (int g = 0; g < 4; ++g) *(v8bf*)&As[buf][ar][ac + g * 8] = rab[g];
  };
#endif
  float rb[8][4];
  auto loadB = [&](int k0) {
    const float* src = w + (size_t)(k0 + bkc) * C_ + n0 + bnc;
#pragma unroll
    for (int j = 0; j < 8; ++j) {
      const float* sj = src + (size_t)j * C_;
#pragma unroll
      for (int i = 0; i < 4; ++i) rb[j][i] = sj[i];
    }
  };
  auto stageB = [&](int buf) {
#pragma unroll
    for (int i = 0; i < 4; ++i) {
      v8bf t;
#pragma unroll
      for (int j = 0; j < 8; ++j) t[j] = f2bf(rb[j][i]);
      *(v8bf*)&Bs[buf][bnc + i][bkc] = t;
    }
  };
  loadA(0, 0); loadB(0); stageB(0);
#ifndef HAVE_ASYNC_LDS
  stageA(0);
#endif
  int cur = 0;
  constexpr int STEPS = C_ / 64;
  for (int s = 0; s < STEPS; ++s) {
#ifdef HAVE_ASYNC_LDS
    __builtin_amdgcn_s_wait_asynccnt(0);
#endif
    __syncthreads();
    if (s + 1 < STEPS) { loadA((s + 1) * 64, cur ^ 1); loadB((s + 1) * 64); }
#pragma unroll
    for (int kc = 0; kc < 2; ++kc) {
      v16bf af[2], bfv[4];
#pragma unroll
      for (int mt = 0; mt < 2; ++mt)
        af[mt] = combine16(
            *(const v8bf*)&As[cur][wm + mt * 16 + l15][kc * 32 + khalf * 8],
            *(const v8bf*)&As[cur][wm + mt * 16 + l15][kc * 32 + 16 + khalf * 8]);
#pragma unroll
      for (int j = 0; j < 4; ++j)
        bfv[j] = combine16(
            *(const v8bf*)&Bs[cur][wn + j * 16 + l15][kc * 32 + khalf * 8],
            *(const v8bf*)&Bs[cur][wn + j * 16 + l15][kc * 32 + 16 + khalf * 8]);
#pragma unroll
      for (int j = 0; j < 4; ++j)
#pragma unroll
        for (int mt = 0; mt < 2; ++mt)
          acc[mt][j] = wmma_bf16(af[mt], bfv[j], acc[mt][j]);
    }
    if (s + 1 < STEPS) {
      stageB(cur ^ 1);
#ifndef HAVE_ASYNC_LDS
      stageA(cur ^ 1);
#endif
      cur ^= 1;
    }
  }
#pragma unroll
  for (int mt = 0; mt < 2; ++mt) {
#pragma unroll
    for (int j = 0; j < 4; ++j) {
#pragma unroll
      for (int i = 0; i < 8; ++i) {
        const int m = m0 + wm + mt * 16 + khalf * 8 + i;
        out[(size_t)m * C_ + n0 + wn + j * 16 + l15] = acc[mt][j][i];
      }
    }
  }
}

// ---- launch ----------------------------------------------------------------

extern "C" void kernel_launch(void* const* d_in, const int* in_sizes, int n_in,
                              void* d_out, int out_size, void* d_ws,
                              size_t ws_size, hipStream_t stream) {
  (void)in_sizes; (void)n_in; (void)out_size; (void)ws_size;
  const float* x = (const float*)d_in[0];
  const float* w_qkv = (const float*)d_in[1];
  const float* w_out = (const float*)d_in[2];
  const int* nv = (const int*)d_in[3];
  __bf16* qkv = (__bf16*)d_ws;
  __bf16* attn = (__bf16*)((char*)d_ws + 3 * QS_ * sizeof(unsigned short));
  float* out = (float*)d_out;

  qkv_gemm_kernel<<<dim3((B_ * T_) / 128, (3 * C_) / 128), 256, 0, stream>>>(
      x, w_qkv, qkv);
  attn_kernel<<<dim3(T_ / 128, H_, B_), 256, 0, stream>>>(qkv, nv, attn);
  out_gemm_kernel<<<dim3((B_ * T_) / 128, C_ / 128), 256, 0, stream>>>(
      attn, w_out, out);
}
